// AttentionConv_27264452395103
// MI455X (gfx1250) — compile-verified, matrix operand order checked by reading, not run on hardware
//
#include <hip/hip_runtime.h>
#include <hip/hip_bf16.h>

typedef __attribute__((ext_vector_type(16))) _Float16 v16h;
typedef __attribute__((ext_vector_type(8)))  float    v8f;

#define CIN    256
#define COUT   256
#define IMG_H  64
#define IMG_W  64
#define TH     4          // output tile rows per block
#define TW     8          // output tile cols per block
#define NPIX   32         // TH*TW
#define HLO_H  6          // TH+2
#define HLO_W  10         // TW+2
#define NHALO  60         // HLO_H*HLO_W
#define NHALO_PAD 64      // padded to 4 WMMA N-tiles
#define LSTR   264        // LDS row stride in f16 (256 + 8 pad; 528B = 33*16)

#define WMAT_ELEMS (COUT * CIN)                    // 65536 per matrix
#define WS_F16_BYTES (3 * WMAT_ELEMS * 2)          // 384 KB

struct alignas(32) PairB128 { int4 lo, hi; };

// ---- WMMA operand builders --------------------------------------------------
// 16-bit A-matrix 16x32 layout (ISA 7.12.2): lanes 0-15 hold K {0..7,16..23},
// lanes 16-31 hold K {8..15,24..31}; f16 index i inside a half maps linearly.
__device__ __forceinline__ v16h load_a_f32(const float* __restrict__ wrow,
                                           int c, bool lowHalf) {
  const int g1 = c * 32 + (lowHalf ? 0 : 8);
  const int g2 = g1 + 16;
  v16h a;
#pragma unroll
  for (int j = 0; j < 8; ++j) a[j]     = (_Float16)wrow[g1 + j];
#pragma unroll
  for (int j = 0; j < 8; ++j) a[j + 8] = (_Float16)wrow[g2 + j];
  return a;
}

// f16 weights (pre-converted into d_ws): two 16B global_load_b128 per chunk.
__device__ __forceinline__ v16h load_a_f16(const _Float16* __restrict__ wrow,
                                           int c, bool lowHalf) {
  const int g1 = c * 32 + (lowHalf ? 0 : 8);
  union { PairB128 raw; v16h v; } u;
  u.raw.lo = *(const int4*)(wrow + g1);
  u.raw.hi = *(const int4*)(wrow + g1 + 16);
  return u.v;
}

// B-matrix 32x16: lane n holds column n with the same K striping as A.
// Rows in LDS are pixel-major with 256 f16 of channels -> two ds_load_b128.
__device__ __forceinline__ v16h load_b_lds(const _Float16* __restrict__ row,
                                           int c, bool lowHalf) {
  const int g1 = c * 32 + (lowHalf ? 0 : 8);
  union { PairB128 raw; v16h v; } u;
  u.raw.lo = *(const int4*)(row + g1);
  u.raw.hi = *(const int4*)(row + g1 + 16);
  return u.v;
}

// ---- prologue: convert wq|wk|wv (f32) -> f16 in workspace -------------------
__global__ __launch_bounds__(256) void convert_weights(
    const float* __restrict__ wq, const float* __restrict__ wk,
    const float* __restrict__ wv, _Float16* __restrict__ dst)
{
  const int idx = blockIdx.x * 256 + threadIdx.x;   // 0 .. 3*65536-1
  const int m = idx >> 16;
  const int e = idx & (WMAT_ELEMS - 1);
  const float* src = (m == 0) ? wq : (m == 1) ? wk : wv;
  dst[idx] = (_Float16)src[e];
}

// ---- fused kernel -----------------------------------------------------------
template <bool F16W>
__global__ __launch_bounds__(256) void attnconv_fused(
    const float* __restrict__ x,  const float* __restrict__ y,
    const float* __restrict__ wq, const float* __restrict__ wk,
    const float* __restrict__ wv, const float* __restrict__ rel_h,
    const float* __restrict__ rel_w, const _Float16* __restrict__ wh,
    float* __restrict__ out)
{
  // pixel-major LDS tiles: [pixel][channel], 16B-aligned rows
  __shared__ __align__(16) _Float16 s_in[NHALO_PAD * LSTR]; // x halo (f16)
  __shared__ __align__(16) _Float16 s_y [NPIX      * LSTR]; // y tile (f16)
  __shared__ __align__(16) _Float16 s_k [NHALO_PAD * LSTR]; // k tile (f16)
  __shared__ __align__(16) _Float16 s_v [NHALO_PAD * LSTR]; // v tile (f16)
  __shared__ float s_rel[768];                               // rel_h|rel_w

  const int tid  = threadIdx.x;
  const int bb   = blockIdx.x >> 7;      // 128 tiles per image
  const int tile = blockIdx.x & 127;
  const int h0   = (tile >> 3) * TH;     // 16 tile-rows
  const int w0   = (tile & 7)  * TW;     // 8 tile-cols

  // ---------------- stage inputs into LDS ----------------
#pragma unroll
  for (int t = tid; t < 4 * CIN; t += 256)   // zero pad halo rows 60..63
    s_in[(NHALO + (t >> 8)) * LSTR + (t & 255)] = (_Float16)0.f;

  for (int jj = tid; jj < HLO_H * CIN; jj += 256) {
    const int hr = jj % HLO_H;
    const int i  = jj / HLO_H;
    const int gh = h0 - 1 + hr;
    const bool rowOK = (gh >= 0) & (gh < IMG_H);
    const float* xrow = x + (((size_t)bb * CIN + i) * IMG_H + (rowOK ? gh : 0)) * IMG_W;
#pragma unroll
    for (int hc = 0; hc < HLO_W; ++hc) {
      const int gw = w0 - 1 + hc;
      const float v = (rowOK && gw >= 0 && gw < IMG_W) ? xrow[gw] : 0.f;
      s_in[(hr * HLO_W + hc) * LSTR + i] = (_Float16)v;
    }
  }

  for (int jj = tid; jj < TH * CIN; jj += 256) {
    const int r = jj & 3;
    const int i = jj >> 2;
    const float* yrow = y + (((size_t)bb * CIN + i) * IMG_H + (h0 + r)) * IMG_W + w0;
#pragma unroll
    for (int c8 = 0; c8 < TW; ++c8)
      s_y[(r * TW + c8) * LSTR + i] = (_Float16)yrow[c8];
  }

  for (int t = tid; t < 768; t += 256)
    s_rel[t] = (t < 384) ? rel_h[t] : rel_w[t - 384];

  __syncthreads();

  const int wave     = tid >> 5;
  const int lane     = tid & 31;
  const int lrow     = lane & 15;
  const bool lowHalf = lane < 16;

  // ---------------- phase 1: k and v GEMMs (WMMA) ----------------
  for (int j = wave; j < 32; j += 8) {         // 4 jobs/wave, uniform
    const int mat = j & 1;
    const int o0  = (j >> 1) * 16;
    v16h A[8];
    if (F16W) {
      const _Float16* wrow = wh + (size_t)(mat + 1) * WMAT_ELEMS + (size_t)(o0 + lrow) * CIN;
#pragma unroll
      for (int c = 0; c < 8; ++c) A[c] = load_a_f16(wrow, c, lowHalf);
    } else {
      const float* wrow = (mat ? wv : wk) + (size_t)(o0 + lrow) * CIN;
#pragma unroll
      for (int c = 0; c < 8; ++c) A[c] = load_a_f32(wrow, c, lowHalf);
    }
    _Float16* dst = mat ? s_v : s_k;
#pragma unroll
    for (int nt = 0; nt < 4; ++nt) {
      v8f acc = {0.f, 0.f, 0.f, 0.f, 0.f, 0.f, 0.f, 0.f};
      const _Float16* brow = &s_in[(nt * 16 + lrow) * LSTR];
#pragma unroll
      for (int c = 0; c < 8; ++c) {
        const v16h B = load_b_lds(brow, c, lowHalf);
        acc = __builtin_amdgcn_wmma_f32_16x16x32_f16(
                  false, A[c], false, B, (short)0, acc, false, false);
      }
      const int px = nt * 16 + lrow;
      const int ob = o0 + (lowHalf ? 0 : 8);
      union { int4 q; _Float16 h[8]; } u;
#pragma unroll
      for (int g = 0; g < 8; ++g) u.h[g] = (_Float16)acc[g];
      *(int4*)&dst[px * LSTR + ob] = u.q;
    }
  }

  __syncthreads();

  // ---------------- phase 2: q GEMM + fused window softmax ----------------
  for (int ot = wave; ot < 16; ot += 8) {      // 2 jobs/wave, uniform
    const int o0 = ot * 16;
    v16h A[8];
    if (F16W) {
      const _Float16* wrow = wh + (size_t)(o0 + lrow) * CIN;     // wq at offset 0
#pragma unroll
      for (int c = 0; c < 8; ++c) A[c] = load_a_f16(wrow, c, lowHalf);
    } else {
      const float* wrow = wq + (size_t)(o0 + lrow) * CIN;
#pragma unroll
      for (int c = 0; c < 8; ++c) A[c] = load_a_f32(wrow, c, lowHalf);
    }

    const int ob = o0 + (lowHalf ? 0 : 8);     // this lane's 8 channels
    const bool useH = (ob < 128);
    float relv[8][3];
#pragma unroll
    for (int g = 0; g < 8; ++g)
#pragma unroll
      for (int t3 = 0; t3 < 3; ++t3)
        relv[g][t3] = useH ? s_rel[(ob + g) * 3 + t3]
                           : s_rel[384 + (ob + g - 128) * 3 + t3];

#pragma unroll
    for (int nt = 0; nt < 2; ++nt) {
      v8f q = {0.f, 0.f, 0.f, 0.f, 0.f, 0.f, 0.f, 0.f};
      const _Float16* brow = &s_y[(nt * 16 + lrow) * LSTR];
#pragma unroll
      for (int c = 0; c < 8; ++c) {
        const v16h B = load_b_lds(brow, c, lowHalf);
        q = __builtin_amdgcn_wmma_f32_16x16x32_f16(
                false, A[c], false, B, (short)0, q, false, false);
      }

      const int px = nt * 16 + lrow;
      const int r  = px >> 3;
      const int cc = px & 7;

      float ssum[8], num[8];
#pragma unroll
      for (int g = 0; g < 8; ++g) { ssum[g] = 0.f; num[g] = 0.f; }

#pragma unroll
      for (int kh = 0; kh < 3; ++kh) {
#pragma unroll
        for (int kw = 0; kw < 3; ++kw) {
          const int hp = (r + kh) * HLO_W + (cc + kw);
          union { int4 q4; _Float16 h[8]; } ku, vu;
          ku.q4 = *(const int4*)&s_k[hp * LSTR + ob];
          vu.q4 = *(const int4*)&s_v[hp * LSTR + ob];
#pragma unroll
          for (int g = 0; g < 8; ++g) {
            const float bias = useH ? relv[g][kh] : relv[g][kw];
            const float e = __expf(q[g] * ((float)ku.h[g] + bias));
            ssum[g] += e;
            num[g]   = fmaf(e, (float)vu.h[g], num[g]);
          }
        }
      }

      const int gh = h0 + r, gw = w0 + cc;
#pragma unroll
      for (int g = 0; g < 8; ++g)
        out[(((size_t)bb * COUT + (ob + g)) * IMG_H + gh) * IMG_W + gw] =
            num[g] / ssum[g];
    }
  }
}

extern "C" void kernel_launch(void* const* d_in, const int* in_sizes, int n_in,
                              void* d_out, int out_size, void* d_ws, size_t ws_size,
                              hipStream_t stream) {
  const float* x     = (const float*)d_in[0];
  const float* y     = (const float*)d_in[1];
  const float* wq    = (const float*)d_in[2];
  const float* wk    = (const float*)d_in[3];
  const float* wv    = (const float*)d_in[4];
  const float* rel_h = (const float*)d_in[5];
  const float* rel_w = (const float*)d_in[6];
  float* out = (float*)d_out;

  const int blocks = 8 * (IMG_H / TH) * (IMG_W / TW);  // 1024

  if (ws_size >= (size_t)WS_F16_BYTES) {
    _Float16* wh = (_Float16*)d_ws;
    convert_weights<<<(3 * WMAT_ELEMS) / 256, 256, 0, stream>>>(wq, wk, wv, wh);
    attnconv_fused<true><<<blocks, 256, 0, stream>>>(
        x, y, wq, wk, wv, rel_h, rel_w, wh, out);
  } else {
    attnconv_fused<false><<<blocks, 256, 0, stream>>>(
        x, y, wq, wk, wv, rel_h, rel_w, nullptr, out);
  }
}
